// TargetUnitHead_18330920419426
// MI455X (gfx1250) — compile-verified, matrix operand order checked by reading, not run on hardware
//
#include <hip/hip_runtime.h>
#include <hip/hip_bf16.h>
#include <math.h>

#define B_     2048
#define E_     512
#define EMB_   256
#define AR_    1024
#define H256_  256
#define H32_   32
#define NT_    256

typedef __attribute__((ext_vector_type(16))) _Float16 v16h;
typedef __attribute__((ext_vector_type(8)))  float    v8f;

// ---- WMMA fragment loaders (wave32 layouts from CDNA5 ISA) ----
// A: 16x32 f16. lane<16: row=lane, K={k0..k0+7} U {k0+16..k0+23};
//               lane>=16: row=lane-16, K shifted by +8.
__device__ __forceinline__ v16h load_a_f32(const float* A, int lda, int row0, int k0, int lane) {
    int r  = row0 + (lane & 15);
    int kb = k0 + ((lane >> 4) << 3);
    const float* p = A + (size_t)r * lda + kb;
    v16h a;
#pragma unroll
    for (int j = 0; j < 8; ++j) { a[j] = (_Float16)p[j]; a[8 + j] = (_Float16)p[16 + j]; }
    return a;
}
__device__ __forceinline__ v16h load_a_i32(const int* A, int lda, int row0, int k0, int lane) {
    int r  = row0 + (lane & 15);
    int kb = k0 + ((lane >> 4) << 3);
    const int* p = A + (size_t)r * lda + kb;
    v16h a;
#pragma unroll
    for (int j = 0; j < 8; ++j) { a[j] = (_Float16)(float)p[j]; a[8 + j] = (_Float16)(float)p[16 + j]; }
    return a;
}
// B: 32x16 f16 (row-major KxN source), column n=lane&15, same K striping as A rows.
__device__ __forceinline__ v16h load_b_f32(const float* Bm, int ldb, int k0, int col0, int lane) {
    int n  = col0 + (lane & 15);
    int kb = k0 + ((lane >> 4) << 3);
    v16h b;
#pragma unroll
    for (int j = 0; j < 8; ++j) {
        b[j]     = (_Float16)Bm[(size_t)(kb + j) * ldb + n];
        b[8 + j] = (_Float16)Bm[(size_t)(kb + 16 + j) * ldb + n];
    }
    return b;
}

// ---- K1: z1 = relu(mask@W_func + b_func) + ar@W_fc1 + b_fc1   (2048x256) ----
__global__ __launch_bounds__(256) void k1_z1(const float* __restrict__ ar,
                                             const int*   __restrict__ mask,
                                             const float* __restrict__ Wfunc,
                                             const float* __restrict__ bfunc,
                                             const float* __restrict__ Wfc1,
                                             const float* __restrict__ bfc1,
                                             float* __restrict__ z1) {
    int wid  = (int)((blockIdx.x * blockDim.x + threadIdx.x) >> 5);
    int lane = (int)(threadIdx.x & 31);
    int mtile = wid >> 4;   // 0..127
    int ntile = wid & 15;   // 0..15
    int row0 = mtile * 16, col0 = ntile * 16;

    v8f accf = {};
    for (int kt = 0; kt < NT_; kt += 32) {
        v16h a = load_a_i32(mask, NT_, row0, kt, lane);
        v16h b = load_b_f32(Wfunc, H256_, kt, col0, lane);
        accf = __builtin_amdgcn_wmma_f32_16x16x32_f16(false, a, false, b, (short)0, accf, false, false);
    }
    v8f accx = {};
    for (int kt = 0; kt < AR_; kt += 32) {
        v16h a = load_a_f32(ar, AR_, row0, kt, lane);
        v16h b = load_b_f32(Wfc1, H256_, kt, col0, lane);
        accx = __builtin_amdgcn_wmma_f32_16x16x32_f16(false, a, false, b, (short)0, accx, false, false);
    }
    int n = col0 + (lane & 15);
    float bf = bfunc[n], bx = bfc1[n];
    int mbase = row0 + ((lane >> 4) << 3);
#pragma unroll
    for (int i = 0; i < 8; ++i) {
        float f = accf[i] + bf;
        f = f > 0.f ? f : 0.f;
        z1[(size_t)(mbase + i) * H256_ + n] = f + accx[i] + bx;
    }
}

// ---- K2a: z2 = z1 @ W_fc2 + b_fc2   (2048x32) ----
__global__ __launch_bounds__(256) void k2a_z2(const float* __restrict__ z1,
                                              const float* __restrict__ Wfc2,
                                              const float* __restrict__ bfc2,
                                              float* __restrict__ z2) {
    int wid  = (int)((blockIdx.x * blockDim.x + threadIdx.x) >> 5);
    int lane = (int)(threadIdx.x & 31);
    int mtile = wid >> 1;   // 0..127
    int ntile = wid & 1;    // 0..1
    int row0 = mtile * 16, col0 = ntile * 16;

    v8f acc = {};
    for (int kt = 0; kt < H256_; kt += 32) {
        v16h a = load_a_f32(z1, H256_, row0, kt, lane);
        v16h b = load_b_f32(Wfc2, H32_, kt, col0, lane);
        acc = __builtin_amdgcn_wmma_f32_16x16x32_f16(false, a, false, b, (short)0, acc, false, false);
    }
    int n = col0 + (lane & 15);
    float bias = bfc2[n];
    int mbase = row0 + ((lane >> 4) << 3);
#pragma unroll
    for (int i = 0; i < 8; ++i)
        z2[(size_t)(mbase + i) * H32_ + n] = acc[i] + bias;
}

// ---- K2b: degenerate LSTM step (h0=c0=0) -> query = sig(o)*tanh(sig(i)*tanh(g)) ----
__device__ __forceinline__ float sigm(float x) { return 1.0f / (1.0f + __expf(-x)); }

__global__ __launch_bounds__(256) void k2b_lstm(const float* __restrict__ z2,
                                                const float* __restrict__ K,
                                                const float* __restrict__ bias,
                                                float* __restrict__ query) {
    int b = (int)(blockIdx.x * blockDim.x + threadIdx.x); // 0..2047
    float x[H32_];
#pragma unroll
    for (int k = 0; k < H32_; ++k) x[k] = z2[(size_t)b * H32_ + k];
    for (int h = 0; h < H32_; ++h) {
        float zi = bias[h], zg = bias[64 + h], zo = bias[96 + h];
#pragma unroll
        for (int k = 0; k < H32_; ++k) {
            const float* kr = K + (size_t)k * 128;
            zi += x[k] * kr[h];
            zg += x[k] * kr[64 + h];
            zo += x[k] * kr[96 + h];
        }
        float c = sigm(zi) * tanhf(zg);
        query[(size_t)b * H32_ + h] = sigm(zo) * tanhf(c);
    }
}

// ---- K3: fold W_conv@query per batch, stream entity once (HBM-bound) ----
__global__ __launch_bounds__(256) void k3_main(const float* __restrict__ entity,
                                               const float* __restrict__ Wconv,
                                               const float* __restrict__ bconv,
                                               const float* __restrict__ query,
                                               const int*   __restrict__ tmask,
                                               float* __restrict__ out) {
    __shared__ float wf[EMB_];
    __shared__ float sc;
    __shared__ float smax[8];
    __shared__ int   sarg[8];

    int b   = (int)blockIdx.x;
    int tid = (int)threadIdx.x;
    const float* qb = query + (size_t)b * H32_;

    // wfold[k] = sum_h W_conv[k,h] * query[b,h]
    {
        const float* wr = Wconv + (size_t)tid * H32_;
        float acc = 0.f;
#pragma unroll
        for (int h = 0; h < H32_; ++h) acc += wr[h] * qb[h];
        wf[tid] = acc;
    }
    if (tid == 0) {
        float c = 0.f;
#pragma unroll
        for (int h = 0; h < H32_; ++h) c += bconv[h] * qb[h];
        sc = c;
    }
    __syncthreads();

    int lane = tid & 31, w = tid >> 5;
    float4 wa = *(const float4*)&wf[lane * 8];
    float4 wb = *(const float4*)&wf[lane * 8 + 4];
    float  cf = sc;
    float  maskf = (float)tmask[b];

    const float* eb = entity + (size_t)b * E_ * EMB_;
    float best = -INFINITY; int bi = 0;

    for (int e = w; e < E_; e += 8) {
        const float* r = eb + (size_t)e * EMB_ + lane * 8;
        float4 r0 = *(const float4*)r;
        float4 r1 = *(const float4*)(r + 4);
        float p = r0.x * wa.x + r0.y * wa.y + r0.z * wa.z + r0.w * wa.w
                + r1.x * wb.x + r1.y * wb.y + r1.z * wb.z + r1.w * wb.w;
#pragma unroll
        for (int off = 16; off > 0; off >>= 1) p += __shfl_xor(p, off, 32);
        if (lane == 0) {
            float y = p + cf;
            out[(size_t)b * E_ + e] = y * maskf;
            if (y > best) { best = y; bi = e; }
        }
    }
    if (lane == 0) { smax[w] = best; sarg[w] = bi; }
    __syncthreads();
    if (tid == 0) {
        float bb = smax[0]; int ba = sarg[0];
#pragma unroll
        for (int i = 1; i < 8; ++i)
            if (smax[i] > bb || (smax[i] == bb && sarg[i] < ba)) { bb = smax[i]; ba = sarg[i]; }
        long long t = (long long)ba * (long long)tmask[b];
        unsigned* ti = (unsigned*)(out + (size_t)B_ * E_);
        ti[(size_t)b * 2 + 0] = (unsigned)((unsigned long long)t & 0xffffffffu);
        ti[(size_t)b * 2 + 1] = (unsigned)((unsigned long long)t >> 32);
    }
}

extern "C" void kernel_launch(void* const* d_in, const int* in_sizes, int n_in,
                              void* d_out, int out_size, void* d_ws, size_t ws_size,
                              hipStream_t stream) {
    const float* ar     = (const float*)d_in[0];   // (2048,1024)
    const int*   mask   = (const int*)  d_in[1];   // (2048,256)
    const int*   tmask  = (const int*)  d_in[2];   // (2048,1)
    const float* entity = (const float*)d_in[3];   // (2048,512,256)
    const float* Wfunc  = (const float*)d_in[4];
    const float* bfunc  = (const float*)d_in[5];
    const float* Wconv  = (const float*)d_in[6];
    const float* bconv  = (const float*)d_in[7];
    const float* Wfc1   = (const float*)d_in[8];
    const float* bfc1   = (const float*)d_in[9];
    const float* Wfc2   = (const float*)d_in[10];
    const float* bfc2   = (const float*)d_in[11];
    const float* lstmK  = (const float*)d_in[12];
    // d_in[13] lstm_recurrent unused (h0 == 0)
    const float* lstmB  = (const float*)d_in[14];

    float* ws    = (float*)d_ws;
    float* z1    = ws;                                   // 2048*256
    float* z2    = z1 + (size_t)B_ * H256_;              // 2048*32
    float* query = z2 + (size_t)B_ * H32_;               // 2048*32

    float* out = (float*)d_out;

    // K1: 2048 wave-tiles (128 M-tiles x 16 N-tiles), 8 waves/block
    k1_z1<<<256, 256, 0, stream>>>(ar, mask, Wfunc, bfunc, Wfc1, bfc1, z1);
    // K2a: 256 wave-tiles (128 x 2)
    k2a_z2<<<32, 256, 0, stream>>>(z1, Wfc2, bfc2, z2);
    // K2b: one thread per batch row
    k2b_lstm<<<8, 256, 0, stream>>>(z2, lstmK, lstmB, query);
    // K3: one block per batch row, streams 512 KiB of entity each (HBM-bound)
    k3_main<<<B_, 256, 0, stream>>>(entity, Wconv, bconv, query, tmask, out);
}
